// BinaryLinear_67156108640875
// MI455X (gfx1250) — matrix operand order repaired and checked by reading, not compiled, not run on hardware
//
#include <hip/hip_runtime.h>
#include <hip/hip_bf16.h>

typedef __attribute__((ext_vector_type(16))) __bf16 v16bf;
typedef __attribute__((ext_vector_type(8)))  __bf16 v8bf;
typedef __attribute__((ext_vector_type(8)))  float  v8f;

#define KDIM 2048      // IN
#define ODIM 2048      // OUT
#define BM 128
#define BN 128
#define BK 64
#define NCHUNK (KDIM / BK)   // 32
#define STR (BK + 8)   // LDS row stride in bf16 elems: 144B, 16B-aligned, conflict-free

// A fragment (16-bit A 16x32 layout, ISA 7.12.2):
// lane L: row M = L%16; elems 0..7 = K[k0..k0+7], elems 8..15 = K[k1..k1+7]
// where k0 = kk + (L/16)*8, k1 = kk + 16 + (L/16)*8.
static __device__ inline v16bf ld_frag_a(const __bf16* rowp, int k0, int k1) {
    v8bf lo = *(const v8bf*)(rowp + k0);
    v8bf hi = *(const v8bf*)(rowp + k1);
    return __builtin_shufflevector(lo, hi, 0,1,2,3,4,5,6,7,8,9,10,11,12,13,14,15);
}

// B fragment (32x16 16-bit B): lane L holds column N = L%16,
// 16 consecutive K values from p; unused (B is loaded with ld_frag_a).
static __device__ inline v16bf ld_frag_b(const __bf16* p) {
    v8bf lo = *(const v8bf*)(p);
    v8bf hi = *(const v8bf*)(p + 8);
    return __builtin_shufflevector(lo, hi, 0,1,2,3,4,5,6,7,8,9,10,11,12,13,14,15);
}

static __device__ inline void load_tiles(const float* __restrict__ X,
                                         const float* __restrict__ W,
                                         int mb, int nb, int kb, int t,
                                         float4 xa[8], float4 wa[8]) {
    const int r0 = t >> 4;      // 0..15
    const int c4 = t & 15;      // 0..15  (float4 column)
#pragma unroll
    for (int rr = 0; rr < 8; ++rr) {
        const int row = r0 + rr * 16;
        xa[rr] = *(const float4*)(X + (size_t)(mb + row) * KDIM + kb + c4 * 4);
        wa[rr] = *(const float4*)(W + (size_t)(nb + row) * KDIM + kb + c4 * 4);
    }
}

static __device__ inline void stage_to_lds(__bf16* sAh, __bf16* sAl, __bf16* sB,
                                           int t, const float4 xa[8], const float4 wa[8]) {
    const int r0 = t >> 4;
    const int c4 = t & 15;
    const __bf16 pone = __builtin_bit_cast(__bf16, (unsigned short)0x3F80);  // +1.0
    const __bf16 none = __builtin_bit_cast(__bf16, (unsigned short)0xBF80);  // -1.0
#pragma unroll
    for (int rr = 0; rr < 8; ++rr) {
        const int row = r0 + rr * 16;
        const float xv[4] = {xa[rr].x, xa[rr].y, xa[rr].z, xa[rr].w};
        const float wv[4] = {wa[rr].x, wa[rr].y, wa[rr].z, wa[rr].w};
#pragma unroll
        for (int e = 0; e < 4; ++e) {
            __bf16 h = (__bf16)xv[e];               // RNE
            __bf16 l = (__bf16)(xv[e] - (float)h);  // residual -> fp32-grade accuracy
            sAh[row * STR + c4 * 4 + e] = h;
            sAl[row * STR + c4 * 4 + e] = l;
            sB [row * STR + c4 * 4 + e] = (wv[e] >= 0.0f) ? pone : none;
        }
    }
}

__global__ __launch_bounds__(256)
void binlin_wmma_kernel(const float* __restrict__ X, const float* __restrict__ W,
                        const float* __restrict__ bias, float* __restrict__ out) {
    // Double-buffered LDS: 2 x (2x16KB A + 16KB B) padded = ~108 KB (WGP has 320 KB)
    __shared__ __bf16 sAh[2][BM * STR];
    __shared__ __bf16 sAl[2][BM * STR];
    __shared__ __bf16 sB [2][BN * STR];

    const int t    = threadIdx.x;
    const int lane = t & 31;
    const int wave = t >> 5;
    const int lm   = lane & 15;
    const int lg   = lane >> 4;
    const int wrow = (wave & 3) * 32;   // 4 waves along M
    const int wcol = (wave >> 2) * 64;  // 2 waves along N
    const int mb   = blockIdx.y * BM;
    const int nb   = blockIdx.x * BN;

    v8f acc[2][4];
    {
        v8f z = {0.f, 0.f, 0.f, 0.f, 0.f, 0.f, 0.f, 0.f};
#pragma unroll
        for (int i = 0; i < 2; ++i)
#pragma unroll
            for (int j = 0; j < 4; ++j) acc[i][j] = z;
    }

    float4 xa[8], wa[8];
    load_tiles(X, W, mb, nb, 0, t, xa, wa);
    stage_to_lds(sAh[0], sAl[0], sB[0], t, xa, wa);
    __syncthreads();

    int p = 0;
    for (int c = 0; c < NCHUNK; ++c) {
        const int kb_next = (c + 1) * BK;
        const bool more = (kb_next < KDIM);
        if (more)
            load_tiles(X, W, mb, nb, kb_next, t, xa, wa);  // global prefetch

        const __bf16* pAh = sAh[p];
        const __bf16* pAl = sAl[p];
        const __bf16* pB  = sB[p];

#pragma unroll
        for (int kk = 0; kk < BK; kk += 32) {
            const int k0 = kk + lg * 8;
            const int k1 = kk + 16 + lg * 8;
            v16bf ah[2], al[2], bf[4];
#pragma unroll
            for (int i = 0; i < 2; ++i) {
                const __bf16* pa = pAh + (size_t)(wrow + i * 16 + lm) * STR;
                const __bf16* pl = pAl + (size_t)(wrow + i * 16 + lm) * STR;
                ah[i] = ld_frag_a(pa, k0, k1);
                al[i] = ld_frag_a(pl, k0, k1);
            }
#pragma unroll
            for (int j = 0; j < 4; ++j)
                bf[j] = ld_frag_a(pB + (size_t)(wcol + j * 16 + lm) * STR, k0, k1);
#pragma unroll
            for (int j = 0; j < 4; ++j) {
#pragma unroll
                for (int i = 0; i < 2; ++i) {
                    acc[i][j] = __builtin_amdgcn_wmma_f32_16x16x32_bf16(
                        false, ah[i], false, bf[j], (short)0, acc[i][j], false, false);
                    acc[i][j] = __builtin_amdgcn_wmma_f32_16x16x32_bf16(
                        false, al[i], false, bf[j], (short)0, acc[i][j], false, false);
                }
            }
        }

        if (more)
            stage_to_lds(sAh[p ^ 1], sAl[p ^ 1], sB[p ^ 1], t, xa, wa);
        __syncthreads();
        p ^= 1;
    }

    // Epilogue: C/D layout (ISA 7.12.2): lane L -> N = L%16, rows r + 8*(L/16).
#pragma unroll
    for (int j = 0; j < 4; ++j) {
        const int col = nb + wcol + j * 16 + lm;
        const float bv = bias[col];
#pragma unroll
        for (int i = 0; i < 2; ++i) {
            const int rbase = mb + wrow + i * 16 + lg * 8;
            v8f cfr = acc[i][j];
#pragma unroll
            for (int r = 0; r < 8; ++r)
                out[(size_t)(rbase + r) * ODIM + col] = cfr[r] + bv;
        }
    }
}

extern "C" void kernel_launch(void* const* d_in, const int* in_sizes, int n_in,
                              void* d_out, int out_size, void* d_ws, size_t ws_size,
                              hipStream_t stream) {
    const float* x = (const float*)d_in[0];   // [N, 2048] fp32
    const float* w = (const float*)d_in[1];   // [2048, 2048] fp32
    const float* b = (const float*)d_in[2];   // [2048] fp32
    float* out = (float*)d_out;               // [N, 2048] fp32

    const int N = in_sizes[0] / KDIM;         // 8192
    dim3 grid(ODIM / BN, N / BM);             // (16, 64)
    binlin_wmma_kernel<<<grid, 256, 0, stream>>>(x, w, b, out);
}
